// SimA_66451734004193
// MI455X (gfx1250) — matrix operand order repaired and checked by reading, not compiled, not run on hardware
//
#include <hip/hip_runtime.h>

// Problem constants (match reference)
#define BDIM  8
#define SEQ   4096
#define CDIM  768
#define HEADS 12
#define HDIM  64
#define MTOK  (BDIM * SEQ)   // 32768 tokens
#define QKVN  (3 * CDIM)     // 2304
#define L1EPS 1e-12f

// GEMM tiling
#define BM 128
#define BN 128
#define BK 32

typedef __attribute__((ext_vector_type(16))) __bf16      bf16x16;
typedef __attribute__((ext_vector_type(8)))  float       floatx8;
typedef __attribute__((ext_vector_type(4)))  unsigned int u32x4;
typedef __attribute__((ext_vector_type(2)))  unsigned int u32x2;
typedef __attribute__((ext_vector_type(4)))  int          i32x4;

union FragAB { u32x4 u[2]; bf16x16 v; };

// gfx1250 async global->LDS path (ASYNCcnt), if the toolchain exposes it.
// Builtin params are int4* (global src, LDS dst), per compiler diagnostic.
#if __has_builtin(__builtin_amdgcn_global_load_async_to_lds_b128) && \
    __has_builtin(__builtin_amdgcn_s_wait_asynccnt)
#define USE_ASYNC 1
typedef __attribute__((address_space(1))) i32x4* gbl_v4i_p;
typedef __attribute__((address_space(3))) i32x4* lds_v4i_p;
#define CP16(dst, src)                                                   \
  __builtin_amdgcn_global_load_async_to_lds_b128(                        \
      (gbl_v4i_p)(src), (lds_v4i_p)(dst), 0, 0)
#endif

__device__ __forceinline__ unsigned short f2bf(float f) {
  unsigned int u = __float_as_uint(f);
  u += 0x7FFFu + ((u >> 16) & 1u);          // round-to-nearest-even
  return (unsigned short)(u >> 16);
}
__device__ __forceinline__ float bf2f(unsigned short b) {
  return __uint_as_float(((unsigned int)b) << 16);
}

// ---------------------------------------------------------------- convert
__global__ void cvt_f32_bf16(const float* __restrict__ in,
                             unsigned short* __restrict__ out, int n) {
  int i = blockIdx.x * blockDim.x + threadIdx.x;
  if (i < n) out[i] = f2bf(in[i]);
}

// ---------------------------------------------------------------- tiled WMMA GEMM
// C[M,N] = A[M,K] @ Bw[N,K]^T  (bf16 bits in, fp32 acc)
// Block tile 128m x 128n, 8 waves (2m x 4n), wave tile 64m x 32n (8 WMMA/K-step),
// double-buffered LDS fed by async global->LDS copies (fallback: sync loads).
template <bool STORE_BF16>
__global__ __launch_bounds__(256)
void wmma_gemm(const unsigned short* __restrict__ A,
               const unsigned short* __restrict__ Bw,
               void* __restrict__ Cout,
               const float* __restrict__ bias,
               int Ndim, int Kdim) {
  __shared__ __align__(16) unsigned short lds_a[2][BM * BK];
  __shared__ __align__(16) unsigned short lds_b[2][BN * BK];

  const int tid  = threadIdx.x;
  const int m0   = blockIdx.y * BM;
  const int n0   = blockIdx.x * BN;
  const int wave = tid >> 5, lane = tid & 31;
  const int wm = wave >> 2, wn = wave & 3;
  const int lrow = lane & 15;
  const int hi   = lane >> 4;               // lane-half select

  floatx8 acc[4][2] = {};

  // Staging map: A and B tiles are both 128 rows x 32 k of bf16 (8 KB each).
  // Thread t stages 16 bf16 (two 16B chunks) of each: row = t>>1, half = t&1.
  const int r = tid >> 1, half = tid & 1;
  const unsigned short* gA = A  + (size_t)(m0 + r) * Kdim + half * 16;
  const unsigned short* gB = Bw + (size_t)(n0 + r) * Kdim + half * 16;
  unsigned short* sA[2] = { &lds_a[0][r * BK + half * 16], &lds_a[1][r * BK + half * 16] };
  unsigned short* sB[2] = { &lds_b[0][r * BK + half * 16], &lds_b[1][r * BK + half * 16] };

#ifdef USE_ASYNC
#define ISSUE_TILE(kk, bsel) do {        \
    CP16(sA[bsel],     gA + (kk));       \
    CP16(sA[bsel] + 8, gA + (kk) + 8);   \
    CP16(sB[bsel],     gB + (kk));       \
    CP16(sB[bsel] + 8, gB + (kk) + 8);   \
  } while (0)
  ISSUE_TILE(0, 0);
#endif

  int buf = 0;
  for (int k0 = 0; k0 < Kdim; k0 += BK) {
#ifdef USE_ASYNC
    if (k0 + BK < Kdim) {
      ISSUE_TILE(k0 + BK, buf ^ 1);                 // prefetch next tile
      __builtin_amdgcn_s_wait_asynccnt(4);          // current tile landed
    } else {
      __builtin_amdgcn_s_wait_asynccnt(0);
    }
    __syncthreads();                                // all waves' data visible
#else
    u32x4 va0 = *(const u32x4*)(gA + k0);
    u32x4 va1 = *(const u32x4*)(gA + k0 + 8);
    u32x4 vb0 = *(const u32x4*)(gB + k0);
    u32x4 vb1 = *(const u32x4*)(gB + k0 + 8);
    __syncthreads();                                // prior reads done (WAR)
    *(u32x4*)sA[buf]       = va0;
    *(u32x4*)(sA[buf] + 8) = va1;
    *(u32x4*)sB[buf]       = vb0;
    *(u32x4*)(sB[buf] + 8) = vb1;
    __syncthreads();
#endif

    FragAB bf[2];
#pragma unroll
    for (int nt = 0; nt < 2; ++nt) {
      const int bbase = (wn * 32 + nt * 16 + lrow) * BK + hi * 16;
      bf[nt].u[0] = *(const u32x4*)&lds_b[buf][bbase];
      bf[nt].u[1] = *(const u32x4*)&lds_b[buf][bbase + 8];
    }
#pragma unroll
    for (int mt = 0; mt < 4; ++mt) {
      FragAB af;
      const int abase = (wm * 64 + mt * 16 + lrow) * BK + hi * 16;
      af.u[0] = *(const u32x4*)&lds_a[buf][abase];
      af.u[1] = *(const u32x4*)&lds_a[buf][abase + 8];
      acc[mt][0] = __builtin_amdgcn_wmma_f32_16x16x32_bf16(
          false, af.v, false, bf[0].v, (short)0, acc[mt][0], false, false);
      acc[mt][1] = __builtin_amdgcn_wmma_f32_16x16x32_bf16(
          false, af.v, false, bf[1].v, (short)0, acc[mt][1], false, false);
    }
    __syncthreads();                                // readers done before overwrite
    buf ^= 1;
  }

#pragma unroll
  for (int mt = 0; mt < 4; ++mt) {
#pragma unroll
    for (int nt = 0; nt < 2; ++nt) {
      const int col = n0 + wn * 32 + nt * 16 + lrow;
#pragma unroll
      for (int i = 0; i < 8; ++i) {
        const int row = m0 + wm * 64 + mt * 16 + hi * 8 + i;
        float v = acc[mt][nt][i];
        if (STORE_BF16) {
          ((unsigned short*)Cout)[(size_t)row * Ndim + col] = f2bf(v);
        } else {
          ((float*)Cout)[(size_t)row * Ndim + col] = v + (bias ? bias[col] : 0.f);
        }
      }
    }
  }
}

// ---------------------------------------------------------------- L1 denominators
// inv[b, d] = 1 / (max(sum_n |q|, eps) * max(sum_n |k|, eps)), d in [0,768)
__global__ __launch_bounds__(256)
void denom_kernel(const unsigned short* __restrict__ qkv,
                  float* __restrict__ inv) {
  const int d = blockIdx.y * 256 + threadIdx.x;
  const int b = blockIdx.x;
  const unsigned short* p = qkv + (size_t)b * SEQ * QKVN;
  float sq = 0.f, sk = 0.f;
  for (int n = 0; n < SEQ; ++n) {
    sq += fabsf(bf2f(p[(size_t)n * QKVN + d]));
    sk += fabsf(bf2f(p[(size_t)n * QKVN + CDIM + d]));
  }
  sq = fmaxf(sq, L1EPS);
  sk = fmaxf(sk, L1EPS);
  inv[b * CDIM + d] = 1.f / (sq * sk);
}

// ---------------------------------------------------------------- kv = k^T v (per b,h)
// One block per (b,h): 16 waves, each owns one 16x16 tile of the 64x64 output.
// k/v tiles transposed through LDS so both WMMA operands read contiguous K.
// Output scaled by inv and stored TRANSPOSED: kv2t[(bh*64 + e)*64 + d].
__global__ __launch_bounds__(512)
void kv_kernel(const unsigned short* __restrict__ qkv,
               const float* __restrict__ inv,
               unsigned short* __restrict__ kv2t) {
  __shared__ __align__(16) unsigned short lds_kT[64 * 32];
  __shared__ __align__(16) unsigned short lds_vT[64 * 32];

  const int bh = blockIdx.x;
  const int b = bh / HEADS, h = bh % HEADS;
  const int tid  = threadIdx.x;
  const int wave = tid >> 5, lane = tid & 31;
  const int dt = wave >> 2, et = wave & 3;
  const int lrow = lane & 15, hi = lane >> 4;

  floatx8 acc = {};
  const int ln  = tid >> 4;         // n within 32-tile
  const int ld0 = (tid & 15) * 4;   // d0 within 64

  for (int n0 = 0; n0 < SEQ; n0 += 32) {
    size_t rowbase = ((size_t)b * SEQ + n0 + ln) * QKVN + h * HDIM + ld0;
    u32x2 kk = *(const u32x2*)(qkv + rowbase + CDIM);       // k slice
    u32x2 vv = *(const u32x2*)(qkv + rowbase + 2 * CDIM);   // v slice
#pragma unroll
    for (int j = 0; j < 2; ++j) {
      unsigned int kw = kk[j], vw = vv[j];
      lds_kT[(ld0 + 2 * j)     * 32 + ln] = (unsigned short)(kw & 0xFFFFu);
      lds_kT[(ld0 + 2 * j + 1) * 32 + ln] = (unsigned short)(kw >> 16);
      lds_vT[(ld0 + 2 * j)     * 32 + ln] = (unsigned short)(vw & 0xFFFFu);
      lds_vT[(ld0 + 2 * j + 1) * 32 + ln] = (unsigned short)(vw >> 16);
    }
    __syncthreads();

    FragAB af, bfg;
    const int drow = dt * 16 + lrow;
    const int ako  = hi * 16;
    af.u[0] = *(const u32x4*)&lds_kT[drow * 32 + ako];
    af.u[1] = *(const u32x4*)&lds_kT[drow * 32 + ako + 8];
    const int ecol = et * 16 + lrow;
    const int bko  = hi * 16;
    bfg.u[0] = *(const u32x4*)&lds_vT[ecol * 32 + bko];
    bfg.u[1] = *(const u32x4*)&lds_vT[ecol * 32 + bko + 8];
    acc = __builtin_amdgcn_wmma_f32_16x16x32_bf16(
        false, af.v, false, bfg.v, (short)0, acc, false, false);
    __syncthreads();
  }

#pragma unroll
  for (int i = 0; i < 8; ++i) {
    int d = dt * 16 + hi * 8 + i;
    int e = et * 16 + lrow;
    float v = acc[i] * inv[b * CDIM + h * HDIM + d];
    kv2t[((size_t)bh * 64 + e) * 64 + d] = f2bf(v);    // transposed store
  }
}

// ---------------------------------------------------------------- out = q_raw @ kv2  (block-diagonal over heads)
__global__ __launch_bounds__(256)
void attn_kernel(const unsigned short* __restrict__ qkv,
                 const unsigned short* __restrict__ kv2t,
                 unsigned short* __restrict__ attn) {
  const int h  = blockIdx.x;
  const int m0 = blockIdx.y * 128;
  const int b  = m0 >> 12;                  // 4096 tokens per batch
  const int bh = b * HEADS + h;
  const int tid  = threadIdx.x;
  const int wave = tid >> 5, lane = tid & 31;
  const int wm = wave >> 2, wn = wave & 3;
  const int lrow = lane & 15, hi = lane >> 4;

  floatx8 acc[4] = {};
#pragma unroll
  for (int k0 = 0; k0 < HDIM; k0 += 32) {
    FragAB bfg;
    const int e = wn * 16 + lrow;
    const unsigned short* gb = kv2t + ((size_t)bh * 64 + e) * 64 + k0 + hi * 16;
    bfg.u[0] = *(const u32x4*)gb;
    bfg.u[1] = *(const u32x4*)(gb + 8);
#pragma unroll
    for (int mt = 0; mt < 4; ++mt) {
      const int m = m0 + wm * 64 + mt * 16 + lrow;
      const unsigned short* ga = qkv + (size_t)m * QKVN + h * HDIM + k0 + hi * 16;
      FragAB af;
      af.u[0] = *(const u32x4*)ga;
      af.u[1] = *(const u32x4*)(ga + 8);
      acc[mt] = __builtin_amdgcn_wmma_f32_16x16x32_bf16(
          false, af.v, false, bfg.v, (short)0, acc[mt], false, false);
    }
  }
  const int col = h * HDIM + wn * 16 + lrow;
#pragma unroll
  for (int mt = 0; mt < 4; ++mt) {
#pragma unroll
    for (int i = 0; i < 8; ++i) {
      int row = m0 + wm * 64 + mt * 16 + hi * 8 + i;
      attn[(size_t)row * CDIM + col] = f2bf(acc[mt][i]);
    }
  }
}

// ---------------------------------------------------------------- launch
extern "C" void kernel_launch(void* const* d_in, const int* in_sizes, int n_in,
                              void* d_out, int out_size, void* d_ws, size_t ws_size,
                              hipStream_t stream) {
  const float* x      = (const float*)d_in[0];
  const float* w_qkv  = (const float*)d_in[1];
  const float* w_proj = (const float*)d_in[2];
  const float* b_proj = (const float*)d_in[3];

  char* ws = (char*)d_ws;
  size_t off = 0;
  unsigned short* xb     = (unsigned short*)(ws + off); off += (size_t)MTOK * CDIM * 2;   // 48 MB
  unsigned short* wqkvb  = (unsigned short*)(ws + off); off += (size_t)QKVN * CDIM * 2;
  unsigned short* wprojb = (unsigned short*)(ws + off); off += (size_t)CDIM * CDIM * 2;
  unsigned short* qkvb   = (unsigned short*)(ws + off); off += (size_t)MTOK * QKVN * 2;   // 144 MB
  float*          inv    = (float*)(ws + off);          off += (size_t)BDIM * CDIM * 4;
  unsigned short* kv2t   = (unsigned short*)(ws + off); off += (size_t)BDIM * HEADS * 64 * 64 * 2;
  unsigned short* attnb  = xb;   // xb is dead after GEMM1 -> reuse for attention output

  (void)in_sizes; (void)n_in; (void)out_size; (void)ws_size;

  const int nx = MTOK * CDIM;
  cvt_f32_bf16<<<(nx + 255) / 256, 256, 0, stream>>>(x, xb, nx);
  const int nq = QKVN * CDIM;
  cvt_f32_bf16<<<(nq + 255) / 256, 256, 0, stream>>>(w_qkv, wqkvb, nq);
  const int np = CDIM * CDIM;
  cvt_f32_bf16<<<(np + 255) / 256, 256, 0, stream>>>(w_proj, wprojb, np);

  // qkv = x @ w_qkv^T   (bf16 out)
  wmma_gemm<true><<<dim3(QKVN / BN, MTOK / BM), 256, 0, stream>>>(
      xb, wqkvb, qkvb, nullptr, QKVN, CDIM);

  // combined L1 normalization scales
  denom_kernel<<<dim3(BDIM, CDIM / 256), 256, 0, stream>>>(qkvb, inv);

  // kv2^T = [ (k^T v) / (denom_q * denom_k) ]^T  per (b,h)
  kv_kernel<<<BDIM * HEADS, 512, 0, stream>>>(qkvb, inv, kv2t);

  // attn = q_raw @ kv2  (block-diagonal over heads, bf16 out)
  attn_kernel<<<dim3(HEADS, MTOK / 128), 256, 0, stream>>>(qkvb, kv2t, attnb);

  // out = attn @ w_proj^T + b_proj  (fp32 out)
  wmma_gemm<false><<<dim3(CDIM / BN, MTOK / BM), 256, 0, stream>>>(
      attnb, wprojb, d_out, b_proj, CDIM, CDIM);
}
